// _GTSComposer_62002147885258
// MI455X (gfx1250) — matrix operand order repaired and checked by reading, not compiled, not run on hardware
//
#include <hip/hip_runtime.h>

// Problem constants (from the reference)
#define BGR   32
#define NPER  128
#define DIM   64
#define EPER  4096
#define TPER  4096
#define NE    (BGR * EPER)      // 131072 edges
#define NT    (BGR * TPER)      // 131072 tokens
#define OUTN  (BGR * NPER * NPER * DIM)  // 33,554,432 floats = 128 MB

typedef __attribute__((ext_vector_type(16))) _Float16 v16h;
typedef __attribute__((ext_vector_type(8)))  float    v8f;
typedef __attribute__((ext_vector_type(4)))  float    f4;

// ---------------------------------------------------------------------------
// Kernel 1: zero-fill the dense output (128 MB; resides in L2 afterwards)
// ---------------------------------------------------------------------------
__global__ void k_zero(f4* __restrict__ out) {
    size_t i = (size_t)blockIdx.x * blockDim.x + threadIdx.x;
    f4 z = {0.f, 0.f, 0.f, 0.f};
    out[i] = z;
}

// ---------------------------------------------------------------------------
// Kernel 2: edge_attr scatter-add.  One thread per (edge, feature).
// out index = src*8192 + (dst&127)*64 + d  (since ptr[g] = 128*g)
// Coalesced reads of edge_attr; atomics hit L2 (output fits in 192 MB L2).
// ---------------------------------------------------------------------------
__global__ void k_edge(const float* __restrict__ ea,
                       const int*   __restrict__ ei,
                       float*       __restrict__ out) {
    int tid = blockIdx.x * blockDim.x + threadIdx.x;   // < NE*64
    int e = tid >> 6;
    int d = tid & 63;
    int s = ei[e];
    int c = ei[NE + e] & 127;
    float v = ea[(size_t)e * 64 + d];
    unsafeAtomicAdd(out + (size_t)s * 8192 + (size_t)c * 64 + d, v);
}

// ---------------------------------------------------------------------------
// Kernel 3: token GEMM (node_proj) via v_wmma_f32_16x16x32_f16 + scatter-add.
// One wave32 = 16 tokens x 64 output features (4 N-tiles x 4 K-chunks).
// ---------------------------------------------------------------------------
__global__ void __launch_bounds__(256)
k_token(const float* __restrict__ x,
        const int*   __restrict__ ti,
        const float* __restrict__ W,
        const float* __restrict__ bias,
        float*       __restrict__ out) {
    const int lane  = threadIdx.x & 31;
    const int wid   = blockIdx.x * (blockDim.x >> 5) + (threadIdx.x >> 5);
    const int tbase = wid * 16;
    const int m     = lane & 15;   // token row within tile (A) / output col (B,C)
    const int hi    = lane >> 4;   // K-half selector per ISA VGPR layout

    // --- Build the four A chunks (K = 0..127 of [x[src] | x[dst]]) ----------
    const int ta = tbase + m;
    const int sa = ti[ta];
    const int da = ti[NT + ta];
    const float* srow = x + (size_t)sa * DIM;
    const float* drow = x + (size_t)da * DIM;

    v16h A[4];
    const int k0 = hi * 8;
#pragma unroll
    for (int c = 0; c < 4; ++c) {
        const float* rowp = (c < 2) ? srow : drow;   // K<64 from src, K>=64 from dst
        const int kk = k0 + ((c & 1) ? 32 : 0);
        const f4* q0 = (const f4*)(rowp + kk);        // K kk..kk+7
        const f4* q1 = (const f4*)(rowp + kk + 16);   // K kk+16..kk+23
        f4 u0 = q0[0], u1 = q0[1], u2 = q1[0], u3 = q1[1];
        v16h a;
        a[0]  = (_Float16)u0.x; a[1]  = (_Float16)u0.y;
        a[2]  = (_Float16)u0.z; a[3]  = (_Float16)u0.w;
        a[4]  = (_Float16)u1.x; a[5]  = (_Float16)u1.y;
        a[6]  = (_Float16)u1.z; a[7]  = (_Float16)u1.w;
        a[8]  = (_Float16)u2.x; a[9]  = (_Float16)u2.y;
        a[10] = (_Float16)u2.z; a[11] = (_Float16)u2.w;
        a[12] = (_Float16)u3.x; a[13] = (_Float16)u3.y;
        a[14] = (_Float16)u3.z; a[15] = (_Float16)u3.w;
        A[c] = a;
    }

    // --- Store-phase addresses: acc[r] belongs to token tbase + hi*8 + r ----
    size_t baseaddr[8];
#pragma unroll
    for (int r = 0; r < 8; ++r) {
        int t  = tbase + hi * 8 + r;
        int s  = ti[t];
        int cc = ti[NT + t] & 127;
        baseaddr[r] = (size_t)s * 8192 + (size_t)cc * 64;
    }

    // --- 4 N-tiles x 4 K-chunks of WMMA, then bias + scatter-add ------------
#pragma unroll
    for (int nt = 0; nt < 4; ++nt) {
        const int n = nt * 16 + m;
        const float* wrow = W + (size_t)n * 128 + hi * 8;
        v8f acc = {};
#pragma unroll
        for (int c = 0; c < 4; ++c) {
            const f4* q = (const f4*)(wrow + c * 32);   // W[n, c*32 + hi*8 + 0..7] and W[n, c*32 + 16 + hi*8 + 0..7]
            f4 w0 = q[0], w1 = q[1], w2 = q[4], w3 = q[5];
            v16h Bm;
            Bm[0]  = (_Float16)w0.x; Bm[1]  = (_Float16)w0.y;
            Bm[2]  = (_Float16)w0.z; Bm[3]  = (_Float16)w0.w;
            Bm[4]  = (_Float16)w1.x; Bm[5]  = (_Float16)w1.y;
            Bm[6]  = (_Float16)w1.z; Bm[7]  = (_Float16)w1.w;
            Bm[8]  = (_Float16)w2.x; Bm[9]  = (_Float16)w2.y;
            Bm[10] = (_Float16)w2.z; Bm[11] = (_Float16)w2.w;
            Bm[12] = (_Float16)w3.x; Bm[13] = (_Float16)w3.y;
            Bm[14] = (_Float16)w3.z; Bm[15] = (_Float16)w3.w;
            acc = __builtin_amdgcn_wmma_f32_16x16x32_f16(
                false, A[c], false, Bm, (short)0, acc, false, false);
        }
        const float bv = bias[n];
#pragma unroll
        for (int r = 0; r < 8; ++r) {
            unsafeAtomicAdd(out + baseaddr[r] + n, acc[r] + bv);
        }
    }
}

// ---------------------------------------------------------------------------
// Host launcher (graph-capture safe: kernels only, all on `stream`)
// ---------------------------------------------------------------------------
extern "C" void kernel_launch(void* const* d_in, const int* in_sizes, int n_in,
                              void* d_out, int out_size, void* d_ws, size_t ws_size,
                              hipStream_t stream) {
    const float* x  = (const float*)d_in[0];   // [4096, 64]
    const int*   ei = (const int*)  d_in[1];   // [2, 131072]
    const float* ea = (const float*)d_in[2];   // [131072, 64]
    // d_in[3] = batch (unused: batch[i] == i/128 by construction)
    const int*   ti = (const int*)  d_in[4];   // [2, 131072]
    const float* W  = (const float*)d_in[5];   // [64, 128]
    const float* b  = (const float*)d_in[6];   // [64]
    float* out = (float*)d_out;                // [32,128,128,64]

    // 1) zero the dense output: 33,554,432 f32 = 8,388,608 float4
    k_zero<<<OUTN / 4 / 256, 256, 0, stream>>>((f4*)out);

    // 2) edge scatter-add: NE*64 = 8,388,608 threads
    k_edge<<<(NE * 64) / 256, 256, 0, stream>>>(ea, ei, out);

    // 3) token GEMM + scatter-add: NT/16 = 8192 waves, 8 waves/block
    k_token<<<NT / 16 / 8, 256, 0, stream>>>(x, ti, W, b, out);
}
